// MultiHeadAttention_1675037246061
// MI455X (gfx1250) — compile-verified, matrix operand order checked by reading, not run on hardware
//
#include <hip/hip_runtime.h>

#define D_MODEL   1024
#define NUM_HEADS 16
#define DEPTH     64
#define SEQ_LEN   2048
#define BATCH     4
#define LDPAD     40            // padded LDS row stride (elems): 80B, 16B-aligned rows

typedef __attribute__((ext_vector_type(16))) __bf16 v16bf;
typedef __attribute__((ext_vector_type(8)))  __bf16 v8bf;
typedef __attribute__((ext_vector_type(8)))  float  v8f;
typedef __attribute__((ext_vector_type(4)))  float  v4f;
typedef int v4i_ __attribute__((vector_size(16)));   // matches builtin's param type

// --------------------------------------------------------------------------
// CDNA5 async global->LDS staging (gfx1250). Signature probed from clang
// diagnostic: arg0 = v4i* (generic/global address), arg1 = LDS destination.
// Guarded: falls back to load+ds_store if the builtin is not exposed.
// --------------------------------------------------------------------------
#if defined(__has_builtin)
#if __has_builtin(__builtin_amdgcn_global_load_async_to_lds_b128)
#define HAVE_ASYNC_LDS 1
#endif
#endif

static __device__ __forceinline__ void async_b128(const void* g, void* l) {
#ifdef HAVE_ASYNC_LDS
  __builtin_amdgcn_global_load_async_to_lds_b128(
      (v4i_*)g, (__attribute__((address_space(3))) v4i_*)l, 0, 0);
#else
  v8bf _v = *(const v8bf*)g;
  *(v8bf*)l = _v;
#endif
}

#ifdef HAVE_ASYNC_LDS
#define ASYNC_WAIT() __asm__ volatile("s_wait_asynccnt 0x0" ::: "memory")
#else
#define ASYNC_WAIT() do {} while (0)
#endif

// --------------------------------------------------------------------------
// Fragment loaders. CDNA5 16-bit A-fragment (16x32, MxK), wave32:
//   lanes 0-15  : row M=lane,    halves i<8 -> K=i,   i>=8 -> K=16+(i-8)
//   lanes 16-31 : row M=lane-16, halves i<8 -> K=8+i, i>=8 -> K=24+(i-8)
// B-fragment (32x16, KxN) mirrors this with lane = column N.
// --------------------------------------------------------------------------
static __device__ __forceinline__ v16bf load_frag_bf16(const __bf16* __restrict__ base,
                                                       int rc, int k0, int ld, int lane) {
  int r  = rc + (lane & 15);
  int b8 = (lane & 16) ? 8 : 0;
  const __bf16* p = base + (size_t)r * ld + k0 + b8;
  v8bf lo = *(const v8bf*)(p);
  v8bf hi = *(const v8bf*)(p + 16);
  v16bf f;
#pragma unroll
  for (int i = 0; i < 8; ++i) { f[i] = lo[i]; f[i + 8] = hi[i]; }
  return f;
}

static __device__ __forceinline__ v16bf lds_frag(const __bf16* p0, int rc, int lane) {
  int r  = rc + (lane & 15);
  int b8 = (lane & 16) ? 8 : 0;
  const __bf16* p = p0 + r * LDPAD + b8;
  v8bf lo = *(const v8bf*)(p);        // ds_load_b128
  v8bf hi = *(const v8bf*)(p + 16);
  v16bf f;
#pragma unroll
  for (int i = 0; i < 8; ++i) { f[i] = lo[i]; f[i + 8] = hi[i]; }
  return f;
}

#define WMMA_BF16(A, B, C)                                                     \
  __builtin_amdgcn_wmma_f32_16x16x32_bf16(false, (A), false, (B), (short)0,    \
                                          (C), false, false)

// --------------------------------------------------------------------------
// One-pass f32 -> bf16 conversion (vectorized 8 elems/thread).
// --------------------------------------------------------------------------
__global__ void __launch_bounds__(256)
cvt_f32_bf16_kernel(const float* __restrict__ in, __bf16* __restrict__ out, int n8) {
  int i = blockIdx.x * 256 + threadIdx.x;
  if (i >= n8) return;
  v4f a = ((const v4f*)in)[2 * i];
  v4f b = ((const v4f*)in)[2 * i + 1];
  v8bf o;
#pragma unroll
  for (int j = 0; j < 4; ++j) { o[j] = (__bf16)a[j]; o[4 + j] = (__bf16)b[j]; }
  ((v8bf*)out)[i] = o;
}

// --------------------------------------------------------------------------
// Tiled GEMM: out = X @ W^T + bias. X:[M,K] bf16, W:[N,K] bf16.
// Block = 256 threads (8 waves, 2x4). Block tile 64x128, wave tile 32x32,
// BK=32, double-buffered LDS staged via async global->LDS b128.
// MODE 0: out bf16 [B,H,S,DEPTH]; MODE 1: out bf16 [B,H,DEPTH,S];
// MODE 2: out f32 row-major [M,N].
// --------------------------------------------------------------------------
template <int MODE>
__global__ void __launch_bounds__(256)
gemm_bf16_wmma(const __bf16* __restrict__ X, const __bf16* __restrict__ W,
               const float* __restrict__ bias, void* __restrict__ out) {
  __shared__ alignas(16) __bf16 As[2][64 * LDPAD];
  __shared__ alignas(16) __bf16 Bs[2][128 * LDPAD];

  const int t    = threadIdx.x;
  const int lane = t & 31;
  const int wave = t >> 5;
  const int wm   = wave & 1;          // 0..1 -> M
  const int wn   = wave >> 1;         // 0..3 -> N
  const int m0   = blockIdx.x * 64;
  const int n0   = blockIdx.y * 128;
  const int K    = D_MODEL;

  // staging assignment: 256 threads, 16B each for A (64x32), 2x16B for B (128x32)
  const int srow = t >> 2;
  const int scol = (t & 3) * 8;

  auto stage = [&](int k0, __bf16* Asb, __bf16* Bsb) {
    async_b128(X + (size_t)(m0 + srow) * K + k0 + scol, Asb + srow * LDPAD + scol);
#pragma unroll
    for (int c = 0; c < 2; ++c) {
      int rb = srow + c * 64;
      async_b128(W + (size_t)(n0 + rb) * K + k0 + scol, Bsb + rb * LDPAD + scol);
    }
  };

  v8f acc[2][2] = {};
  const int nk = K / 32;

  stage(0, As[0], Bs[0]);
  for (int ks = 0; ks < nk; ++ks) {
    ASYNC_WAIT();
    __syncthreads();                  // staged data visible; prev readers done
    const int cur = ks & 1;
    if (ks + 1 < nk) stage((ks + 1) * 32, As[cur ^ 1], Bs[cur ^ 1]);

    const __bf16* Asb = As[cur];
    const __bf16* Bsb = Bs[cur];
    v16bf a0 = lds_frag(Asb, wm * 32 + 0,  lane);
    v16bf a1 = lds_frag(Asb, wm * 32 + 16, lane);
    v16bf b0 = lds_frag(Bsb, wn * 32 + 0,  lane);
    v16bf b1 = lds_frag(Bsb, wn * 32 + 16, lane);
    acc[0][0] = WMMA_BF16(a0, b0, acc[0][0]);
    acc[0][1] = WMMA_BF16(a0, b1, acc[0][1]);
    acc[1][0] = WMMA_BF16(a1, b0, acc[1][0]);
    acc[1][1] = WMMA_BF16(a1, b1, acc[1][1]);
  }

  const int hi8 = (lane & 16) ? 8 : 0;
#pragma unroll
  for (int im = 0; im < 2; ++im) {
#pragma unroll
    for (int in = 0; in < 2; ++in) {
      int n = n0 + wn * 32 + in * 16 + (lane & 15);
      float bv = bias[n];
      int h = n / DEPTH, dch = n % DEPTH;
#pragma unroll
      for (int r = 0; r < 8; ++r) {
        int m = m0 + wm * 32 + im * 16 + r + hi8;
        float val = acc[im][in][r] + bv;
        if (MODE == 2) {
          ((float*)out)[(size_t)m * D_MODEL + n] = val;
        } else {
          int bidx = m >> 11, s = m & (SEQ_LEN - 1);
          size_t idx = (MODE == 0)
              ? ((((size_t)bidx * NUM_HEADS + h) * SEQ_LEN + s) * DEPTH + dch)
              : ((((size_t)bidx * NUM_HEADS + h) * DEPTH + dch) * SEQ_LEN + s);
          ((__bf16*)out)[idx] = (__bf16)val;
        }
      }
    }
  }
}

// --------------------------------------------------------------------------
// Causal flash attention. 4 waves/block; each wave owns a 16-row query tile:
// QK^T (4 WMMA) -> online softmax (shfl_xor row reductions within each
// 16-lane half) -> P relayout through per-wave LDS (s_wait_dscnt, no block
// barrier since waves have different causal trip counts) -> PV (4 WMMA).
// --------------------------------------------------------------------------
__global__ void __launch_bounds__(128)
flash_attn_kernel(const __bf16* __restrict__ Qh,   // [B,H,S,DEPTH]
                  const __bf16* __restrict__ Kh,   // [B,H,S,DEPTH]
                  const __bf16* __restrict__ Vt,   // [B,H,DEPTH,S]
                  __bf16* __restrict__ Out) {      // [B,S,D_MODEL]
  __shared__ alignas(16) __bf16 ptile[4][16 * 32];
  const int lane = threadIdx.x & 31;
  const int wave = threadIdx.x >> 5;
  const int qblocks = SEQ_LEN / 64;
  int qb = blockIdx.x % qblocks;
  int bh = blockIdx.x / qblocks;
  int h  = bh % NUM_HEADS;
  int b  = bh / NUM_HEADS;

  const __bf16* Q  = Qh + (size_t)bh * SEQ_LEN * DEPTH;
  const __bf16* Kp = Kh + (size_t)bh * SEQ_LEN * DEPTH;
  const __bf16* Vp = Vt + (size_t)bh * DEPTH * SEQ_LEN;

  const int q0 = qb * 64 + wave * 16;
  const float scale = 0.125f;                       // 1/sqrt(64)
  const int hi8  = (lane & 16) ? 8 : 0;
  const int ncol = lane & 15;

  v16bf aq0 = load_frag_bf16(Q, q0, 0,  DEPTH, lane);
  v16bf aq1 = load_frag_bf16(Q, q0, 32, DEPTH, lane);

  v8f o0 = {}, o1 = {}, o2 = {}, o3 = {};
  float mrun[8], lrun[8];
#pragma unroll
  for (int r = 0; r < 8; ++r) { mrun[r] = -3.0e38f; lrun[r] = 0.0f; }

  __bf16* pt = &ptile[wave][0];
  const int nkb = (q0 + 16 + 31) / 32;
  for (int kb = 0; kb < nkb; ++kb) {
    const int kbase = kb * 32;
    if (kb + 1 < nkb)
      __builtin_prefetch(Kp + (size_t)(kbase + 32) * DEPTH, 0, 1);

    v8f s[2];
#pragma unroll
    for (int nt = 0; nt < 2; ++nt) {
      v8f a = {};
      v16bf bk0 = load_frag_bf16(Kp, kbase + nt * 16, 0,  DEPTH, lane);
      v16bf bk1 = load_frag_bf16(Kp, kbase + nt * 16, 32, DEPTH, lane);
      a = WMMA_BF16(aq0, bk0, a);
      a = WMMA_BF16(aq1, bk1, a);
      s[nt] = a;
    }

    float p0[8], p1[8], alpha[8];
#pragma unroll
    for (int r = 0; r < 8; ++r) {
      int qrow = q0 + r + hi8;
      float v0 = s[0][r] * scale; if (kbase + ncol      > qrow) v0 = -1.0e9f;
      float v1 = s[1][r] * scale; if (kbase + 16 + ncol > qrow) v1 = -1.0e9f;
      float tmax = fmaxf(v0, v1);
#pragma unroll
      for (int msk = 1; msk < 16; msk <<= 1)
        tmax = fmaxf(tmax, __shfl_xor(tmax, msk, 32));
      float mnew = fmaxf(mrun[r], tmax);
      float a  = __expf(mrun[r] - mnew);
      float e0 = __expf(v0 - mnew);
      float e1 = __expf(v1 - mnew);
      float ts = e0 + e1;
#pragma unroll
      for (int msk = 1; msk < 16; msk <<= 1)
        ts += __shfl_xor(ts, msk, 32);
      lrun[r] = lrun[r] * a + ts;
      mrun[r] = mnew;
      alpha[r] = a; p0[r] = e0; p1[r] = e1;
    }

#pragma unroll
    for (int r = 0; r < 8; ++r) {
      o0[r] *= alpha[r]; o1[r] *= alpha[r]; o2[r] *= alpha[r]; o3[r] *= alpha[r];
      int row = r + hi8;
      pt[row * 32 + ncol]      = (__bf16)p0[r];
      pt[row * 32 + 16 + ncol] = (__bf16)p1[r];
    }
    __asm__ volatile("s_wait_dscnt 0x0" ::: "memory");

    v16bf pa;
    {
      int b8 = (lane & 16) ? 8 : 0;
      const __bf16* p = pt + (lane & 15) * 32 + b8;
      v8bf lo = *(const v8bf*)(p);
      v8bf hi = *(const v8bf*)(p + 16);
#pragma unroll
      for (int i = 0; i < 8; ++i) { pa[i] = lo[i]; pa[i + 8] = hi[i]; }
    }

    o0 = WMMA_BF16(pa, load_frag_bf16(Vp, 0,  kbase, SEQ_LEN, lane), o0);
    o1 = WMMA_BF16(pa, load_frag_bf16(Vp, 16, kbase, SEQ_LEN, lane), o1);
    o2 = WMMA_BF16(pa, load_frag_bf16(Vp, 32, kbase, SEQ_LEN, lane), o2);
    o3 = WMMA_BF16(pa, load_frag_bf16(Vp, 48, kbase, SEQ_LEN, lane), o3);
  }

  __bf16* outp = Out + (size_t)b * SEQ_LEN * D_MODEL + (size_t)h * DEPTH;
#pragma unroll
  for (int r = 0; r < 8; ++r) {
    float inv = 1.0f / lrun[r];
    __bf16* op = outp + (size_t)(q0 + r + hi8) * D_MODEL;
    op[ncol]      = (__bf16)(o0[r] * inv);
    op[16 + ncol] = (__bf16)(o1[r] * inv);
    op[32 + ncol] = (__bf16)(o2[r] * inv);
    op[48 + ncol] = (__bf16)(o3[r] * inv);
  }
}

// --------------------------------------------------------------------------
extern "C" void kernel_launch(void* const* d_in, const int* in_sizes, int n_in,
                              void* d_out, int out_size, void* d_ws, size_t ws_size,
                              hipStream_t stream) {
  const float* q       = (const float*)d_in[0];
  const float* k       = (const float*)d_in[1];
  const float* v       = (const float*)d_in[2];
  const float* wq_w    = (const float*)d_in[3];
  const float* wq_b    = (const float*)d_in[4];
  const float* wk_w    = (const float*)d_in[5];
  const float* wk_b    = (const float*)d_in[6];
  const float* wv_w    = (const float*)d_in[7];
  const float* wv_b    = (const float*)d_in[8];
  const float* dense_w = (const float*)d_in[9];
  const float* dense_b = (const float*)d_in[10];

  const size_t nAct = (size_t)BATCH * SEQ_LEN * D_MODEL;   // 8 Mi elems
  const size_t nW   = (size_t)D_MODEL * D_MODEL;           // 1 Mi elems

  char* ws = (char*)d_ws;
  size_t off = 0;
  auto carve = [&](size_t elems) { char* p = ws + off; off += elems * 2; return (__bf16*)p; };
  __bf16* Xq = carve(nAct);
  __bf16* Xk = carve(nAct);
  __bf16* Xv = carve(nAct);
  __bf16* Wq = carve(nW);
  __bf16* Wk = carve(nW);
  __bf16* Wv = carve(nW);
  __bf16* Wd = carve(nW);
  __bf16* Qh = carve(nAct);
  __bf16* Kh = carve(nAct);
  __bf16* Vt = carve(nAct);
  __bf16* Ao = carve(nAct);

  // one-pass f32 -> bf16
  const int actB = (int)(nAct / 8 / 256);   // 4096 blocks
  const int wB   = (int)(nW / 8 / 256);     // 512 blocks
  cvt_f32_bf16_kernel<<<actB, 256, 0, stream>>>(q, Xq, (int)(nAct / 8));
  cvt_f32_bf16_kernel<<<actB, 256, 0, stream>>>(k, Xk, (int)(nAct / 8));
  cvt_f32_bf16_kernel<<<actB, 256, 0, stream>>>(v, Xv, (int)(nAct / 8));
  cvt_f32_bf16_kernel<<<wB, 256, 0, stream>>>(wq_w, Wq, (int)(nW / 8));
  cvt_f32_bf16_kernel<<<wB, 256, 0, stream>>>(wk_w, Wk, (int)(nW / 8));
  cvt_f32_bf16_kernel<<<wB, 256, 0, stream>>>(wv_w, Wv, (int)(nW / 8));
  cvt_f32_bf16_kernel<<<wB, 256, 0, stream>>>(dense_w, Wd, (int)(nW / 8));

  const int M = BATCH * SEQ_LEN;            // 8192
  dim3 grid(M / 64, D_MODEL / 128);         // (128, 8)
  gemm_bf16_wmma<0><<<grid, 256, 0, stream>>>(Xq, Wq, wq_b, Qh);
  gemm_bf16_wmma<0><<<grid, 256, 0, stream>>>(Xk, Wk, wk_b, Kh);
  gemm_bf16_wmma<1><<<grid, 256, 0, stream>>>(Xv, Wv, wv_b, Vt);

  flash_attn_kernel<<<BATCH * NUM_HEADS * (SEQ_LEN / 64), 128, 0, stream>>>(Qh, Kh, Vt, Ao);

  gemm_bf16_wmma<2><<<grid, 256, 0, stream>>>(Ao, Wd, dense_b, (float*)d_out);
}
